// StackedMamba_17102559773098
// MI455X (gfx1250) — compile-verified, matrix operand order checked by reading, not run on hardware
//
#include <hip/hip_runtime.h>
#include <hip/hip_bf16.h>

// ---------------- model constants ----------------
#define D_MODEL   1024
#define D_INNER   2048
#define HEADDIM   64
#define NHEADS    32
#define D_STATE   128
#define KCONV     4
#define CONV_DIM  (D_INNER + 2*D_STATE)            // 2304
#define D_IN_PROJ (2*D_INNER + 2*D_STATE + NHEADS) // 4384
#define NUM_BLOCKS 4
#define BATCH     4
#define SEQ       2048
#define ROWS      (BATCH*SEQ)                      // 8192
#define LABEL     1000
#define EPS_F     1e-5f

typedef _Float16 v16h __attribute__((ext_vector_type(16)));
typedef _Float16 h8   __attribute__((ext_vector_type(8)));
typedef float    v8f  __attribute__((ext_vector_type(8)));

// ---------------- helpers ----------------
__device__ __forceinline__ float sigmoidf_(float x) { return 1.0f / (1.0f + __expf(-x)); }
__device__ __forceinline__ float siluf_(float x)    { return x * sigmoidf_(x); }

__device__ __forceinline__ float block_reduce_sum(float v, float* red) {
    int tid = threadIdx.x;
    red[tid] = v;
    __syncthreads();
    #pragma unroll
    for (int s = 128; s > 0; s >>= 1) {
        if (tid < s) red[tid] += red[tid + s];
        __syncthreads();
    }
    float r = red[0];
    __syncthreads();
    return r;
}

// async 16B global -> LDS copy (ASYNCcnt path), and its fence
__device__ __forceinline__ void async_copy_b128(unsigned lds_off, const void* gaddr) {
    asm volatile("global_load_async_to_lds_b128 %0, %1, off"
                 :: "v"(lds_off), "v"(gaddr)
                 : "memory");
}
__device__ __forceinline__ void wait_asynccnt0() {
    asm volatile("s_wait_asynccnt 0x0" ::: "memory");
}

// ---------------- embedding gather ----------------
__global__ __launch_bounds__(256) void k_embed(const int* __restrict__ tok,
                                               const float* __restrict__ emb,
                                               float* __restrict__ x) {
    size_t r = blockIdx.x;
    int tid = threadIdx.x;
    size_t t = (size_t)tok[r];
    #pragma unroll
    for (int i = 0; i < 4; ++i) {
        int c = tid + i * 256;
        x[r * D_MODEL + c] = emb[t * D_MODEL + c];
    }
}

// ---------------- f32 -> f16 convert (grid-stride) ----------------
__global__ __launch_bounds__(256) void k_cvt_f16(const float* __restrict__ in,
                                                 _Float16* __restrict__ out,
                                                 long long n) {
    long long stride = (long long)gridDim.x * 256;
    for (long long i = (long long)blockIdx.x * 256 + threadIdx.x; i < n; i += stride)
        out[i] = (_Float16)in[i];
}

// ---------------- layernorm row of D_MODEL -> f16 ----------------
__global__ __launch_bounds__(256) void k_layernorm(const float* __restrict__ x,
                                                   const float* __restrict__ w,
                                                   const float* __restrict__ b,
                                                   _Float16* __restrict__ out) {
    __shared__ float red[256];
    size_t r = blockIdx.x;
    int tid = threadIdx.x;
    float v[4];
    float s = 0.0f;
    #pragma unroll
    for (int i = 0; i < 4; ++i) {
        v[i] = x[r * D_MODEL + tid + i * 256];
        s += v[i];
    }
    float mu = block_reduce_sum(s, red) * (1.0f / D_MODEL);
    float q = 0.0f;
    #pragma unroll
    for (int i = 0; i < 4; ++i) { float d = v[i] - mu; q += d * d; }
    float var = block_reduce_sum(q, red) * (1.0f / D_MODEL);
    float inv = rsqrtf(var + EPS_F);
    #pragma unroll
    for (int i = 0; i < 4; ++i) {
        int c = tid + i * 256;
        out[r * D_MODEL + c] = (_Float16)((v[i] - mu) * inv * w[c] + b[c]);
    }
}

// ---------------- WMMA fragment load from global (16-bit A layout) ----------
// lane<16 : K chunks {k..k+7, k+16..k+23}; lane>=16 : {k+8..k+15, k+24..k+31}
__device__ __forceinline__ v16h load_frag(const _Float16* __restrict__ p, int ld,
                                          int row, int k, int half, int l15,
                                          int maxRow) {
    int r = row + l15;
    r = (r < maxRow) ? r : maxRow;
    const _Float16* q = p + (size_t)r * ld + k + half * 8;
    v16h f;
    *(h8*)&f       = *(const h8*)(q);         // global_load_b128
    *((h8*)&f + 1) = *(const h8*)(q + 16);
    return f;
}

// WMMA fragment load from an LDS tile stored row-major [rows][32] halves
__device__ __forceinline__ v16h lds_frag(const _Float16* base, int row,
                                         int half, int l15) {
    const _Float16* q = base + (size_t)(row + l15) * 32 + half * 8;
    v16h f;
    *(h8*)&f       = *(const h8*)(q);         // ds_load_b128
    *((h8*)&f + 1) = *(const h8*)(q + 16);
    return f;
}

// ---------------- f16 WMMA GEMM: C[M,N] = A[M,K] * Bw[N,K]^T (+Res)(+bias) ---
// 256 threads = 8 waves; block tile 128(M)x64(N); wave tile 32x32 (2x2 frags).
// B slab (64x32 f16 = 4KB) staged in LDS via async copies, double buffered:
// copy of step k+32 overlaps WMMAs of step k; fenced by s_wait_asynccnt+barrier.
__global__ __launch_bounds__(256)
void k_gemm_f16(const _Float16* __restrict__ A, const _Float16* __restrict__ Bw,
                const float* __restrict__ Res, const float* __restrict__ bias,
                float* __restrict__ C, int M, int N, int K,
                int lda, int ldb, int ldc) {
    __shared__ _Float16 Bsh[2][64][32];       // 8 KB double buffer

    const int tid  = threadIdx.x;
    const int lane = tid & 31;
    const int wave = tid >> 5;
    const int m0 = blockIdx.x * 128 + (wave & 3) * 32;
    const int nblk = blockIdx.y * 64;
    const int nW = (wave >> 2) * 32;          // wave's N offset inside block tile
    const int half = lane >> 4;
    const int l15 = lane & 15;

    // per-thread async-copy assignment: 64 rows x 4 chunks of 16B
    const int crow = tid >> 2;
    const int cchk = (tid & 3) * 8;           // halves
    int grow = nblk + crow;
    grow = (grow < N - 1) ? grow : (N - 1);   // clamp ragged N edge
    const _Float16* gB = Bw + (size_t)grow * ldb + cchk;

    v8f acc[2][2];
    #pragma unroll
    for (int mi = 0; mi < 2; ++mi)
        #pragma unroll
        for (int ni = 0; ni < 2; ++ni)
            #pragma unroll
            for (int j = 0; j < 8; ++j) {
                int mm = m0 + mi * 16 + half * 8 + j;
                int nn = nblk + nW + ni * 16 + l15;
                acc[mi][ni][j] = (Res && nn < N) ? Res[(size_t)mm * ldc + nn] : 0.0f;
            }

    // prologue: fill buffer 0 with K-slab 0
    async_copy_b128((unsigned)(size_t)&Bsh[0][crow][cchk], gB);
    wait_asynccnt0();
    __syncthreads();

    int buf = 0;
    for (int k = 0; k < K; k += 32) {
        if (k + 32 < K)  // overlap next slab copy with this step's WMMAs
            async_copy_b128((unsigned)(size_t)&Bsh[buf ^ 1][crow][cchk], gB + k + 32);

        __builtin_prefetch(A + (size_t)(m0 + l15) * lda + k + 32, 0, 3);

        v16h a0 = load_frag(A, lda, m0,      k, half, l15, M - 1);
        v16h a1 = load_frag(A, lda, m0 + 16, k, half, l15, M - 1);
        v16h b0 = lds_frag(&Bsh[buf][0][0], nW,      half, l15);
        v16h b1 = lds_frag(&Bsh[buf][0][0], nW + 16, half, l15);

        acc[0][0] = __builtin_amdgcn_wmma_f32_16x16x32_f16(false, a0, false, b0,
                        (short)0, acc[0][0], false, false);
        acc[0][1] = __builtin_amdgcn_wmma_f32_16x16x32_f16(false, a0, false, b1,
                        (short)0, acc[0][1], false, false);
        acc[1][0] = __builtin_amdgcn_wmma_f32_16x16x32_f16(false, a1, false, b0,
                        (short)0, acc[1][0], false, false);
        acc[1][1] = __builtin_amdgcn_wmma_f32_16x16x32_f16(false, a1, false, b1,
                        (short)0, acc[1][1], false, false);

        wait_asynccnt0();   // own wave's copy into buf^1 landed in LDS
        __syncthreads();    // all waves done copying & done reading buf
        buf ^= 1;
    }

    #pragma unroll
    for (int mi = 0; mi < 2; ++mi)
        #pragma unroll
        for (int ni = 0; ni < 2; ++ni)
            #pragma unroll
            for (int j = 0; j < 8; ++j) {
                int mm = m0 + mi * 16 + half * 8 + j;
                int nn = nblk + nW + ni * 16 + l15;
                if (nn < N) {
                    float v = acc[mi][ni][j];
                    if (bias) v += bias[nn];
                    C[(size_t)mm * ldc + nn] = v;
                }
            }
}

// ---------------- dt = softplus(raw + bias); dA = exp(-exp(A_log)*dt) --------
__global__ __launch_bounds__(256) void k_dt(const float* __restrict__ zx,
                                            const float* __restrict__ dt_bias,
                                            const float* __restrict__ A_log,
                                            float* __restrict__ dt,
                                            float* __restrict__ dA, int total) {
    int i = blockIdx.x * 256 + threadIdx.x;
    if (i >= total) return;
    int h = i & (NHEADS - 1);
    size_t r = (size_t)(i >> 5);
    float xr = zx[r * D_IN_PROJ + (D_INNER + CONV_DIM) + h] + dt_bias[h];
    float dtv = (xr > 20.0f) ? xr : log1pf(__expf(xr));
    dt[i] = dtv;
    dA[i] = __expf(-__expf(A_log[h]) * dtv);
}

// ---------------- depthwise causal conv (K=4) + bias + silu ------------------
__global__ __launch_bounds__(256) void k_conv(const float* __restrict__ zx,
                                              const float* __restrict__ cw,
                                              const float* __restrict__ cb,
                                              float* __restrict__ xbc) {
    size_t r = blockIdx.x;                 // b*SEQ + l
    int l = (int)(r & (SEQ - 1));
    int tid = threadIdx.x;
    for (int c = tid; c < CONV_DIM; c += 256) {
        float acc = cb[c];
        #pragma unroll
        for (int k = 0; k < KCONV; ++k) {
            int lp = l + k - (KCONV - 1);
            if (lp >= 0) {
                long long rr = (long long)r + k - (KCONV - 1);
                acc += zx[rr * D_IN_PROJ + D_INNER + c] * cw[c * KCONV + k];
            }
        }
        xbc[r * CONV_DIM + c] = siluf_(acc);
    }
}

// ---------------- selective scan: one workgroup per (b,h) -------------------
__global__ __launch_bounds__(256) void k_scan(const float* __restrict__ xbc,
                                              const float* __restrict__ dt,
                                              const float* __restrict__ dA,
                                              const float* __restrict__ Dp,
                                              float* __restrict__ y) {
    __shared__ float Bs[D_STATE];
    __shared__ float Cs[D_STATE];
    int bh = blockIdx.x;
    int b = bh >> 5, h = bh & (NHEADS - 1);
    int tid = threadIdx.x;
    int p = tid >> 2;
    int nq = (tid & 3) * 32;

    float hreg[32];
    #pragma unroll
    for (int i = 0; i < 32; ++i) hreg[i] = 0.0f;
    float Dv = Dp[h];

    for (int t = 0; t < SEQ; ++t) {
        size_t r = (size_t)b * SEQ + t;
        const float* row = xbc + r * CONV_DIM;
        __syncthreads();
        if (tid < D_STATE)       Bs[tid] = row[D_INNER + tid];
        else                     Cs[tid - D_STATE] = row[D_INNER + D_STATE + (tid - D_STATE)];
        __syncthreads();
        float dAv  = dA[r * NHEADS + h];
        float dtv  = dt[r * NHEADS + h];
        float xv   = row[h * HEADDIM + p];
        float coef = dtv * xv;
        float acc = 0.0f;
        #pragma unroll
        for (int i = 0; i < 32; ++i) {
            float hv = fmaf(hreg[i], dAv, coef * Bs[nq + i]);
            hreg[i] = hv;
            acc = fmaf(hv, Cs[nq + i], acc);
        }
        acc += __shfl_xor(acc, 1, 32);
        acc += __shfl_xor(acc, 2, 32);
        if ((tid & 3) == 0)
            y[r * D_INNER + h * HEADDIM + p] = acc + Dv * xv;
    }
}

// ---------------- y * silu(z), RMS norm, -> f16 ------------------------------
__global__ __launch_bounds__(256) void k_gate_rms(const float* __restrict__ y,
                                                  const float* __restrict__ zx,
                                                  const float* __restrict__ rms_w,
                                                  _Float16* __restrict__ yh) {
    __shared__ float red[256];
    size_t r = blockIdx.x;
    int tid = threadIdx.x;
    float v[8];
    float ss = 0.0f;
    #pragma unroll
    for (int i = 0; i < 8; ++i) {
        int c = tid + i * 256;
        float zv = zx[r * D_IN_PROJ + c];
        float yv = y[r * D_INNER + c] * siluf_(zv);
        v[i] = yv;
        ss += yv * yv;
    }
    float scale = rsqrtf(block_reduce_sum(ss, red) * (1.0f / D_INNER) + EPS_F);
    #pragma unroll
    for (int i = 0; i < 8; ++i) {
        int c = tid + i * 256;
        yh[r * D_INNER + c] = (_Float16)(v[i] * scale * rms_w[c]);
    }
}

// ---------------- launcher ----------------
extern "C" void kernel_launch(void* const* d_in, const int* in_sizes, int n_in,
                              void* d_out, int out_size, void* d_ws, size_t ws_size,
                              hipStream_t stream) {
    const int*   tokens  = (const int*)  d_in[0];
    const float* emb     = (const float*)d_in[1];
    const float* ln_w    = (const float*)d_in[2];
    const float* ln_b    = (const float*)d_in[3];
    const float* in_W    = (const float*)d_in[4];
    const float* conv_w  = (const float*)d_in[5];
    const float* conv_b  = (const float*)d_in[6];
    const float* dt_bias = (const float*)d_in[7];
    const float* A_log   = (const float*)d_in[8];
    const float* Dp      = (const float*)d_in[9];
    const float* rms_w   = (const float*)d_in[10];
    const float* out_W   = (const float*)d_in[11];
    const float* lin_W   = (const float*)d_in[12];
    const float* lin_b   = (const float*)d_in[13];
    float* out = (float*)d_out;

    // workspace carve-up (bytes)
    char* ws = (char*)d_ws;
    float*    x    = (float*)   (ws);                      // 33,554,432
    _Float16* xnh  = (_Float16*)(ws + 33554432ULL);        // 16,777,216
    float*    zx   = (float*)   (ws + 50331648ULL);        // 143,654,912
    float*    xbc  = (float*)   (ws + 193986560ULL);       // 75,497,472
    float*    dtb  = (float*)   (ws + 269484032ULL);       // 1,048,576
    float*    dAb  = (float*)   (ws + 270532608ULL);       // 1,048,576
    float*    yb   = (float*)   (ws + 271581184ULL);       // 67,108,864
    _Float16* yh   = (_Float16*)(ws + 338690048ULL);       // 33,554,432
    _Float16* wh   = (_Float16*)(ws + 372244480ULL);       // 8,978,432

    // 1) embedding
    k_embed<<<ROWS, 256, 0, stream>>>(tokens, emb, x);

    // 2) mamba blocks
    for (int bi = 0; bi < NUM_BLOCKS; ++bi) {
        const float* lw = ln_w    + (size_t)bi * D_MODEL;
        const float* lb = ln_b    + (size_t)bi * D_MODEL;
        const float* iW = in_W    + (size_t)bi * D_IN_PROJ * D_MODEL;
        const float* cw = conv_w  + (size_t)bi * CONV_DIM * KCONV;
        const float* cb = conv_b  + (size_t)bi * CONV_DIM;
        const float* db = dt_bias + (size_t)bi * NHEADS;
        const float* al = A_log   + (size_t)bi * NHEADS;
        const float* dp = Dp      + (size_t)bi * NHEADS;
        const float* rw = rms_w   + (size_t)bi * D_INNER;
        const float* oW = out_W   + (size_t)bi * D_MODEL * D_INNER;

        k_layernorm<<<ROWS, 256, 0, stream>>>(x, lw, lb, xnh);

        k_cvt_f16<<<2048, 256, 0, stream>>>(iW, wh, (long long)D_IN_PROJ * D_MODEL);
        {
            dim3 g(ROWS / 128, (D_IN_PROJ + 63) / 64);
            k_gemm_f16<<<g, 256, 0, stream>>>(xnh, wh, nullptr, nullptr, zx,
                                              ROWS, D_IN_PROJ, D_MODEL,
                                              D_MODEL, D_MODEL, D_IN_PROJ);
        }

        k_dt<<<(ROWS * NHEADS + 255) / 256, 256, 0, stream>>>(zx, db, al, dtb, dAb,
                                                              ROWS * NHEADS);
        k_conv<<<ROWS, 256, 0, stream>>>(zx, cw, cb, xbc);
        k_scan<<<BATCH * NHEADS, 256, 0, stream>>>(xbc, dtb, dAb, dp, yb);
        k_gate_rms<<<ROWS, 256, 0, stream>>>(yb, zx, rw, yh);

        k_cvt_f16<<<2048, 256, 0, stream>>>(oW, wh, (long long)D_MODEL * D_INNER);
        {
            dim3 g(ROWS / 128, (D_MODEL + 63) / 64);
            // residual fused: x = x + yh * oW^T
            k_gemm_f16<<<g, 256, 0, stream>>>(yh, wh, x, nullptr, x,
                                              ROWS, D_MODEL, D_INNER,
                                              D_INNER, D_INNER, D_MODEL);
        }
    }

    // 3) head: logits = x * lin_W^T + lin_b
    k_cvt_f16<<<2048, 256, 0, stream>>>(x, xnh, (long long)ROWS * D_MODEL);
    k_cvt_f16<<<2048, 256, 0, stream>>>(lin_W, wh, (long long)LABEL * D_MODEL);
    {
        dim3 g(ROWS / 128, (LABEL + 63) / 64);
        k_gemm_f16<<<g, 256, 0, stream>>>(xnh, wh, nullptr, lin_b, out,
                                          ROWS, LABEL, D_MODEL,
                                          D_MODEL, D_MODEL, LABEL);
    }
}
